// GNN_Merged_Mesh_Enhanced_14181982011843
// MI455X (gfx1250) — matrix operand order repaired and checked by reading, not compiled, not run on hardware
//
#include <hip/hip_runtime.h>
#include <hip/hip_bf16.h>
#include <stdint.h>

typedef __attribute__((ext_vector_type(16))) _Float16 v16h;
typedef __attribute__((ext_vector_type(8)))  _Float16 v8h;
typedef __attribute__((ext_vector_type(8)))  float    v8f;

#define NH      15   // per-layer hidden (messages)
#define NGLOB   3
#define NMASK   10
#define COUT    20   // fc2 output
#define NGRAPH  8

__device__ __forceinline__ float prelu(float x, float a) { return x >= 0.f ? x : a * x; }

// ---------------------------------------------------------------------------
// Embedding MLP: 3 -> 30 (PReLU) -> 10 (PReLU). One thread per node.
// ---------------------------------------------------------------------------
__global__ void k_embed(const float* __restrict__ in,
                        const float* __restrict__ W1, const float* __restrict__ b1,
                        const float* __restrict__ a1p,
                        const float* __restrict__ W2, const float* __restrict__ b2,
                        const float* __restrict__ a2p,
                        float* __restrict__ out, int N) {
  int n = blockIdx.x * blockDim.x + threadIdx.x;
  if (n >= N) return;
  float a1 = *a1p, a2 = *a2p;
  float x0 = in[n * 3 + 0], x1 = in[n * 3 + 1], x2 = in[n * 3 + 2];
  float h[30];
#pragma unroll
  for (int j = 0; j < 30; ++j)
    h[j] = prelu(b1[j] + W1[j * 3 + 0] * x0 + W1[j * 3 + 1] * x1 + W1[j * 3 + 2] * x2, a1);
#pragma unroll
  for (int j = 0; j < 10; ++j) {
    float s = b2[j];
#pragma unroll
    for (int k = 0; k < 30; ++k) s += W2[j * 30 + k] * h[k];
    out[n * 10 + j] = prelu(s, a2);
  }
}

// ---------------------------------------------------------------------------
// Counting kernels
// ---------------------------------------------------------------------------
__global__ void k_count(const int* __restrict__ idx, float* __restrict__ cnt, int n) {
  int i = blockIdx.x * blockDim.x + threadIdx.x;
  if (i < n) atomicAdd(&cnt[idx[i]], 1.0f);
}

__global__ void k_counti(const int* __restrict__ idx, int* __restrict__ cnt, int n) {
  int i = blockIdx.x * blockDim.x + threadIdx.x;
  if (i < n) atomicAdd(&cnt[idx[i]], 1);
}

// ---------------------------------------------------------------------------
// Single-block chunked exclusive scan (degrees -> CSR offsets, off[N] = E)
// ---------------------------------------------------------------------------
__global__ void k_scan(const int* __restrict__ deg, int* __restrict__ off, int N) {
  __shared__ int tmp[2][256];
  __shared__ int carry;
  if (threadIdx.x == 0) carry = 0;
  __syncthreads();
  for (int base = 0; base < N; base += 256) {
    int i = base + threadIdx.x;
    int v = (i < N) ? deg[i] : 0;
    int orig = v;
    tmp[0][threadIdx.x] = v;
    __syncthreads();
    int pin = 0;
    for (int s = 1; s < 256; s <<= 1) {
      int pout = pin ^ 1;
      int t = tmp[pin][threadIdx.x];
      if (threadIdx.x >= s) t += tmp[pin][threadIdx.x - s];
      tmp[pout][threadIdx.x] = t;
      __syncthreads();
      pin = pout;
    }
    if (i < N) off[i] = carry + tmp[pin][threadIdx.x] - orig;  // exclusive
    __syncthreads();
    if (threadIdx.x == 255) carry += tmp[pin][255];
    __syncthreads();
  }
  if (threadIdx.x == 0) off[N] = carry;
}

// Scatter src ids into dst-sorted order (CSR adjacency), rebuilt every call
__global__ void k_scatter(const int* __restrict__ src, const int* __restrict__ dst,
                          const int* __restrict__ off, int* __restrict__ cursor,
                          int* __restrict__ esrc, int E) {
  int e = blockIdx.x * blockDim.x + threadIdx.x;
  if (e >= E) return;
  int d = dst[e];
  int p = off[d] + atomicAdd(&cursor[d], 1);
  esrc[p] = src[e];
}

// ---------------------------------------------------------------------------
// fglobal: p = PReLU(h @ Wg^T + bg), atomic-accumulate per graph
// ---------------------------------------------------------------------------
__global__ void k_global(const float* __restrict__ h,
                         const float* __restrict__ Wg, const float* __restrict__ bg,
                         const float* __restrict__ a3p, const int* __restrict__ batch,
                         float* __restrict__ gsum, int N, int cin) {
  int n = blockIdx.x * blockDim.x + threadIdx.x;
  if (n >= N) return;
  float a3 = *a3p;
  int b = batch[n];
  for (int j = 0; j < NGLOB; ++j) {
    float s = bg[j];
    for (int k = 0; k < cin; ++k) s += Wg[j * cin + k] * h[(size_t)n * cin + k];
    atomicAdd(&gsum[b * NGLOB + j], prelu(s, a3));
  }
}

__global__ void k_gdiv(const float* __restrict__ gsum, const float* __restrict__ gcnt,
                       float* __restrict__ gavg) {
  int i = threadIdx.x;
  if (i < NGRAPH * NGLOB) gavg[i] = gsum[i] / fmaxf(gcnt[i / NGLOB], 1.0f);
}

// ---------------------------------------------------------------------------
// feat = [h(cin), mask_emb(10), gpr(3)]   stride 48 (f32 — gather source)
// ---------------------------------------------------------------------------
__global__ void k_feat(const float* __restrict__ h, const float* __restrict__ maskE,
                       const float* __restrict__ gavg, const int* __restrict__ batch,
                       float* __restrict__ feat, int N, int cin) {
  int n = blockIdx.x * blockDim.x + threadIdx.x;
  if (n >= N) return;
  float* f = feat + (size_t)n * 48;
  for (int k = 0; k < cin; ++k)   f[k]            = h[(size_t)n * cin + k];
  for (int k = 0; k < NMASK; ++k) f[cin + k]      = maskE[(size_t)n * NMASK + k];
  int b = batch[n];
  for (int k = 0; k < NGLOB; ++k) f[cin + NMASK + k] = gavg[b * NGLOB + k];
}

// ---------------------------------------------------------------------------
// Dst-centric edge kernel (CSR): for each node, walk incoming edges, compute
// e = PReLU(We*rel+be), m = PReLU(W1*[feat[src], e]+b1), accumulate mean in
// registers. Zero atomics. Weights staged in LDS; prefetch next gather row.
// ---------------------------------------------------------------------------
template <int CIN>
__global__ void __launch_bounds__(256)
k_edges_csr(const int* __restrict__ off, const int* __restrict__ esrc,
            const float* __restrict__ mn, const float* __restrict__ feat,
            const float* __restrict__ W1, const float* __restrict__ b1,
            const float* __restrict__ We, const float* __restrict__ be,
            const float* __restrict__ a1p, const float* __restrict__ a4p,
            float* __restrict__ agg, int N) {
  constexpr int FW = CIN + NMASK + NGLOB;  // gathered feature width
  constexpr int KW = FW + NH;              // fc1 reduction dim
  __shared__ float sW1[NH * KW];
  __shared__ float sb1[NH];
  __shared__ float sWe[NH * 3];
  __shared__ float sbe[NH];
  for (int i = threadIdx.x; i < NH * KW; i += blockDim.x) sW1[i] = W1[i];
  for (int i = threadIdx.x; i < NH * 3;  i += blockDim.x) sWe[i] = We[i];
  for (int i = threadIdx.x; i < NH;      i += blockDim.x) { sb1[i] = b1[i]; sbe[i] = be[i]; }
  __syncthreads();
  int n = blockIdx.x * blockDim.x + threadIdx.x;
  if (n >= N) return;
  float a1 = *a1p, a4 = *a4p;
  int e0 = off[n], e1 = off[n + 1];
  float d0 = mn[n * 3 + 0], d1 = mn[n * 3 + 1], d2 = mn[n * 3 + 2];
  float acc[NH];
#pragma unroll
  for (int j = 0; j < NH; ++j) acc[j] = 0.f;
  for (int p = e0; p < e1; ++p) {
    int s = esrc[p];
    if (p + 1 < e1) __builtin_prefetch(&feat[(size_t)esrc[p + 1] * 48], 0, 0);
    float r0 = d0 - mn[s * 3 + 0];
    float r1 = d1 - mn[s * 3 + 1];
    float r2 = d2 - mn[s * 3 + 2];
    float ev[NH];
#pragma unroll
    for (int j = 0; j < NH; ++j)
      ev[j] = prelu(sbe[j] + sWe[j * 3] * r0 + sWe[j * 3 + 1] * r1 + sWe[j * 3 + 2] * r2, a4);
    float fv[FW];
    const float* fp = feat + (size_t)s * 48;
#pragma unroll
    for (int k = 0; k < FW; ++k) fv[k] = fp[k];
#pragma unroll
    for (int j = 0; j < NH; ++j) {
      float m = sb1[j];
#pragma unroll
      for (int k = 0; k < FW; ++k) m += sW1[j * KW + k] * fv[k];
#pragma unroll
      for (int k = 0; k < NH; ++k) m += sW1[j * KW + FW + k] * ev[k];
      acc[j] += prelu(m, a1);
    }
  }
  float inv = 1.0f / fmaxf((float)(e1 - e0), 1.0f);
#pragma unroll
  for (int j = 0; j < NH; ++j) agg[(size_t)n * NH + j] = acc[j] * inv;
}

// ---------------------------------------------------------------------------
// Stage fc2 input [h, mask, agg, gpr] as fp16, stride 64 (K padded)
// ---------------------------------------------------------------------------
__global__ void k_fc2pre(const float* __restrict__ h, const float* __restrict__ maskE,
                         const float* __restrict__ agg,
                         const float* __restrict__ gavg, const int* __restrict__ batch,
                         _Float16* __restrict__ in2, int N, int cin) {
  int n = blockIdx.x * blockDim.x + threadIdx.x;
  if (n >= N) return;
  _Float16* o = in2 + (size_t)n * 64;
  int b = batch[n];
  int p = 0;
  for (int k = 0; k < cin; ++k)   o[p++] = (_Float16)h[(size_t)n * cin + k];
  for (int k = 0; k < NMASK; ++k) o[p++] = (_Float16)maskE[(size_t)n * NMASK + k];
  for (int k = 0; k < NH; ++k)    o[p++] = (_Float16)agg[(size_t)n * NH + k];
  for (int k = 0; k < NGLOB; ++k) o[p++] = (_Float16)gavg[b * NGLOB + k];
  for (; p < 64; ++p)             o[p]   = (_Float16)0.0f;
}

// ---------------------------------------------------------------------------
// fc2 via WMMA: out = PReLU(in2 @ W2^T + b2) (+skip). One wave per 16-node
// tile; 2 N-tiles x 2 K-steps of v_wmma_f32_16x16x32_f16, f32 accumulate.
// Weights zero-padded to 32x64 fp16 in LDS. Skip is compile-time; full tiles
// take a branchless epilogue.
// ---------------------------------------------------------------------------
template <bool HAS_SKIP>
__global__ void __launch_bounds__(256)
k_fc2(const _Float16* __restrict__ in2,
      const float* __restrict__ W2, const float* __restrict__ b2,
      const float* __restrict__ a2p, const float* __restrict__ skip,
      float* __restrict__ out, int N, int Kact) {
  __shared__ __align__(16) _Float16 sW[32 * 64];
  __shared__ float sb[32];
  for (int i = threadIdx.x; i < 32 * 64; i += blockDim.x) {
    int r = i >> 6, k = i & 63;
    sW[i] = (r < COUT && k < Kact) ? (_Float16)W2[r * Kact + k] : (_Float16)0.0f;
  }
  for (int i = threadIdx.x; i < 32; i += blockDim.x) sb[i] = (i < COUT) ? b2[i] : 0.0f;
  __syncthreads();

  int lane = threadIdx.x & 31;
  int wv   = threadIdx.x >> 5;
  int ntiles = (N + 15) >> 4;
  int tile = blockIdx.x * 8 + wv;
  if (tile >= ntiles) return;      // wave-uniform; EXEC stays full for WMMA
  int m0 = tile << 4;
  bool full = (m0 + 16 <= N);      // wave-uniform

  float a2 = *a2p;
  int r  = lane & 15;
  int hh = lane >> 4;              // half-wave selector
  int mrow = m0 + r; if (mrow > N - 1) mrow = N - 1;
  // A (16x32 f16): lane holds row m=r; VGPR0-3 -> K = hh*8 + 0..7, VGPR4-7 -> K = 16 + hh*8 + 0..7
  const _Float16* ap = in2 + (size_t)mrow * 64 + hh * 8;

#pragma unroll
  for (int nt = 0; nt < 2; ++nt) {
    int n0 = nt * 16;
    // B (32x16 f16): lane holds column n; K = hh*8 + 0..7, 16 + hh*8 + 0..7 (matches A)
    const _Float16* bp = sW + (size_t)(n0 + r) * 64 + hh * 8;
    v8f c = {};
#pragma unroll
    for (int ks = 0; ks < 64; ks += 32) {
      v8h alo = *(const v8h*)(ap + ks);
      v8h ahi = *(const v8h*)(ap + ks + 16);
      v8h blo = *(const v8h*)(bp + ks);
      v8h bhi = *(const v8h*)(bp + ks + 16);
      v16h a, b;
#pragma unroll
      for (int i = 0; i < 8; ++i) {
        a[i] = alo[i]; a[8 + i] = ahi[i];
        b[i] = blo[i]; b[8 + i] = bhi[i];
      }
      c = __builtin_amdgcn_wmma_f32_16x16x32_f16(false, a, false, b, (short)0, c, false, false);
    }
    int ncol = n0 + r;
    if (ncol < COUT) {
      float bias = sb[ncol];
      if (full) {
#pragma unroll
        for (int v = 0; v < 8; ++v) {
          int row = m0 + v + hh * 8;   // D layout: VGPR v -> M = v (+8 for upper half-wave)
          float val = prelu(c[v] + bias, a2);
          if (HAS_SKIP) val += skip[(size_t)row * COUT + ncol];
          out[(size_t)row * COUT + ncol] = val;
        }
      } else {
#pragma unroll
        for (int v = 0; v < 8; ++v) {
          int row = m0 + v + hh * 8;
          if (row < N) {
            float val = prelu(c[v] + bias, a2);
            if (HAS_SKIP) val += skip[(size_t)row * COUT + ncol];
            out[(size_t)row * COUT + ncol] = val;
          }
        }
      }
    }
  }
}

// ---------------------------------------------------------------------------
// Graph mean-pool of final node features + prediction head
// ---------------------------------------------------------------------------
__global__ void k_pool(const float* __restrict__ h, const int* __restrict__ batch,
                       float* __restrict__ gp, int N) {
  int n = blockIdx.x * blockDim.x + threadIdx.x;
  if (n >= N) return;
  int b = batch[n];
  for (int j = 0; j < COUT; ++j) atomicAdd(&gp[b * COUT + j], h[(size_t)n * COUT + j]);
}

__global__ void k_pred(const float* __restrict__ gp, const float* __restrict__ gcnt,
                       const float* __restrict__ W1, const float* __restrict__ b1,
                       const float* __restrict__ ap,
                       const float* __restrict__ W2, const float* __restrict__ b2,
                       float* __restrict__ out) {
  int g = threadIdx.x;
  if (g >= NGRAPH) return;
  float a = *ap;
  float inv = 1.0f / fmaxf(gcnt[g], 1.0f);
  float v[COUT], h[COUT];
  for (int k = 0; k < COUT; ++k) v[k] = gp[g * COUT + k] * inv;
  for (int j = 0; j < COUT; ++j) {
    float s = b1[j];
    for (int k = 0; k < COUT; ++k) s += W1[j * COUT + k] * v[k];
    h[j] = prelu(s, a);
  }
  for (int j = 0; j < 2; ++j) {
    float s = b2[j];
    for (int k = 0; k < COUT; ++k) s += W2[j * COUT + k] * h[k];
    out[g * 2 + j] = 10.0f * s;
  }
}

// ---------------------------------------------------------------------------
// Host orchestration
// ---------------------------------------------------------------------------
extern "C" void kernel_launch(void* const* d_in, const int* in_sizes, int n_in,
                              void* d_out, int out_size, void* d_ws, size_t ws_size,
                              hipStream_t stream) {
  (void)n_in; (void)out_size; (void)ws_size;
  const float* x     = (const float*)d_in[0];
  const float* mask  = (const float*)d_in[1];
  const int*   ae    = (const int*)d_in[2];
  const float* mn    = (const float*)d_in[3];
  const int*   batch = (const int*)d_in[4];
  int N = in_sizes[0] / 3;
  int E = in_sizes[2] / 2;
  const int* src = ae;
  const int* dst = ae + E;

  auto F = [&](int i) { return (const float*)d_in[i]; };
  // params flattened in jax pytree (sorted-key) order, after 6 positional inputs
  int pi = 6;
  const float *em_a1 = F(pi + 0), *em_a2 = F(pi + 1), *em_W1 = F(pi + 2),
              *em_b1 = F(pi + 3), *em_W2 = F(pi + 4), *em_b2 = F(pi + 5);
  pi += 6;
  const float *mk_a1 = F(pi + 0), *mk_a2 = F(pi + 1), *mk_W1 = F(pi + 2),
              *mk_b1 = F(pi + 3), *mk_W2 = F(pi + 4), *mk_b2 = F(pi + 5);
  pi += 6;
  struct LP { const float *a1, *a2, *a3, *a4, *fc1W, *fc1b, *fc2W, *fc2b, *feW, *feb, *fgW, *fgb; } L[10];
  for (int i = 0; i < 10; ++i) {
    L[i].a1 = F(pi + 0); L[i].a2 = F(pi + 1); L[i].a3 = F(pi + 2); L[i].a4 = F(pi + 3);
    L[i].fc1W = F(pi + 4); L[i].fc1b = F(pi + 5);
    L[i].fc2W = F(pi + 6); L[i].fc2b = F(pi + 7);
    L[i].feW  = F(pi + 8); L[i].feb  = F(pi + 9);
    L[i].fgW  = F(pi + 10); L[i].fgb = F(pi + 11);
    pi += 12;
  }
  const float *pr_a = F(pi + 0), *pr_W1 = F(pi + 1), *pr_b1 = F(pi + 2),
              *pr_W2 = F(pi + 3), *pr_b2 = F(pi + 4);

  // workspace carve (≈78 MB)
  uintptr_t base = (uintptr_t)d_ws;
  auto alloc = [&](size_t bytes) -> void* {
    void* p = (void*)base;
    base += (bytes + 255) & ~(size_t)255;
    return p;
  };
  float*    xe   = (float*)alloc((size_t)N * 10 * 4);
  float*    me   = (float*)alloc((size_t)N * 10 * 4);
  float*    buf1 = (float*)alloc((size_t)N * COUT * 4);
  float*    buf2 = (float*)alloc((size_t)N * COUT * 4);
  float*    buf3 = (float*)alloc((size_t)N * COUT * 4);
  float*    feat = (float*)alloc((size_t)N * 48 * 4);
  _Float16* in2  = (_Float16*)alloc((size_t)N * 64 * 2);
  float*    agg  = (float*)alloc((size_t)N * NH * 4);
  int*      ideg = (int*)alloc((size_t)N * 4);
  int*      off  = (int*)alloc((size_t)(N + 1) * 4);
  int*      curs = (int*)alloc((size_t)N * 4);
  int*      esrc = (int*)alloc((size_t)E * 4);
  float*    gcnt = (float*)alloc(32 * 4);
  float*    gsum = (float*)alloc(32 * 4);
  float*    gavg = (float*)alloc(32 * 4);
  float*    gp   = (float*)alloc(256 * 4);

  const int tb = 256;
  int gb  = (N + tb - 1) / tb;
  int gbe = (E + tb - 1) / tb;

  // --- one-time (per call) prep: embeddings + CSR build (graph is static) ---
  hipMemsetAsync(ideg, 0, (size_t)N * 4, stream);
  hipMemsetAsync(curs, 0, (size_t)N * 4, stream);
  hipMemsetAsync(gcnt, 0, 32 * 4, stream);
  k_embed<<<gb, tb, 0, stream>>>(x, em_W1, em_b1, em_a1, em_W2, em_b2, em_a2, xe, N);
  k_embed<<<gb, tb, 0, stream>>>(mask, mk_W1, mk_b1, mk_a1, mk_W2, mk_b2, mk_a2, me, N);
  k_counti<<<gbe, tb, 0, stream>>>(dst, ideg, E);
  k_count<<<gb, tb, 0, stream>>>(batch, gcnt, N);
  k_scan<<<1, 256, 0, stream>>>(ideg, off, N);
  k_scatter<<<gbe, tb, 0, stream>>>(src, dst, off, curs, esrc, E);

  // layer schedule with skip connections (o1/o2/o3 all land in buf1)
  float*       outb[10]  = {buf1, buf2, buf3, buf1, buf2, buf3, buf1, buf2, buf3, buf1};
  const float* skipb[10] = {nullptr, nullptr, buf1, nullptr, nullptr, buf1, nullptr, nullptr, buf1, nullptr};

  const float* cur = xe;
  int ntiles = (N + 15) / 16;
  int gb2 = (ntiles + 7) / 8;
  for (int i = 0; i < 10; ++i) {
    int cin = (i == 0) ? 10 : 20;
    hipMemsetAsync(gsum, 0, 32 * 4, stream);
    k_global<<<gb, tb, 0, stream>>>(cur, L[i].fgW, L[i].fgb, L[i].a3, batch, gsum, N, cin);
    k_gdiv<<<1, 32, 0, stream>>>(gsum, gcnt, gavg);
    k_feat<<<gb, tb, 0, stream>>>(cur, me, gavg, batch, feat, N, cin);
    if (cin == 10)
      k_edges_csr<10><<<gb, tb, 0, stream>>>(off, esrc, mn, feat, L[i].fc1W, L[i].fc1b,
                                             L[i].feW, L[i].feb, L[i].a1, L[i].a4, agg, N);
    else
      k_edges_csr<20><<<gb, tb, 0, stream>>>(off, esrc, mn, feat, L[i].fc1W, L[i].fc1b,
                                             L[i].feW, L[i].feb, L[i].a1, L[i].a4, agg, N);
    k_fc2pre<<<gb, tb, 0, stream>>>(cur, me, agg, gavg, batch, in2, N, cin);
    if (skipb[i])
      k_fc2<true><<<gb2, 256, 0, stream>>>(in2, L[i].fc2W, L[i].fc2b, L[i].a2, skipb[i],
                                           outb[i], N, cin + 28);
    else
      k_fc2<false><<<gb2, 256, 0, stream>>>(in2, L[i].fc2W, L[i].fc2b, L[i].a2, nullptr,
                                            outb[i], N, cin + 28);
    cur = outb[i];
  }

  hipMemsetAsync(gp, 0, 256 * 4, stream);
  k_pool<<<gb, tb, 0, stream>>>(cur, batch, gp, N);
  k_pred<<<1, 32, 0, stream>>>(gp, gcnt, pr_W1, pr_b1, pr_a, pr_W2, pr_b2, (float*)d_out);
}
